// SGCAFEModule_85822036509009
// MI455X (gfx1250) — matrix operand order repaired and checked by reading, not compiled, not run on hardware
//
#include <hip/hip_runtime.h>
#include <hip/hip_bf16.h>

typedef __attribute__((ext_vector_type(16))) _Float16 v16h;
typedef __attribute__((ext_vector_type(8)))  _Float16 v8h;
typedef __attribute__((ext_vector_type(8)))  float    v8f;

#define CLIP 1024
#define INNER 256
#define NHEAD 4
#define HDIM 64
#define NB 16
#define NN 1024
#define NROWS (NB * NN)        // 16384

static __device__ __forceinline__ v16h load_afrag(const _Float16* rowp, int k0, int half) {
    // A (MxK row-major): lane row = lane%16; 16 f16 = two contiguous 8-halfword chunks
    v16h a;
    ((v8h*)&a)[0] = *(const v8h*)(rowp + k0 + half * 8);
    ((v8h*)&a)[1] = *(const v8h*)(rowp + k0 + 16 + half * 8);
    return a;
}

static __device__ __forceinline__ v16h load_bfrag(const _Float16* colp, int k0, int half) {
    // B stored transposed (BT[col][k]): lane col = lane%16; two 8-halfword chunks at k0+half*8 and k0+16+half*8
    v16h b;
    ((v8h*)&b)[0] = *(const v8h*)(colp + k0 + half * 8);
    ((v8h*)&b)[1] = *(const v8h*)(colp + k0 + 16 + half * 8);
    return b;
}

#define WMMA_F16(A, B, C) \
    __builtin_amdgcn_wmma_f32_16x16x32_f16(false, (A), false, (B), (short)0, (C), false, false)

// 16x64 strip GEMM with explicit double buffering: next k-step's fragments are
// loaded into fresh registers before the current k-step's WMMAs so loads stay
// in flight behind the matrix pipe (restores per-wave load/WMMA overlap).
static __device__ __forceinline__ void gemm_strip4(
    const _Float16* __restrict__ Arow,
    const _Float16* __restrict__ B0, const _Float16* __restrict__ B1,
    const _Float16* __restrict__ B2, const _Float16* __restrict__ B3,
    int nkt, int half, v8f acc[4]) {
    v16h af  = load_afrag(Arow, 0, half);
    v16h bf0 = load_bfrag(B0, 0, half);
    v16h bf1 = load_bfrag(B1, 0, half);
    v16h bf2 = load_bfrag(B2, 0, half);
    v16h bf3 = load_bfrag(B3, 0, half);
    for (int kt = 0; kt < nkt - 1; ++kt) {
        const int k0 = (kt + 1) * 32;
        v16h afn = load_afrag(Arow, k0, half);
        v16h bn0 = load_bfrag(B0, k0, half);
        v16h bn1 = load_bfrag(B1, k0, half);
        v16h bn2 = load_bfrag(B2, k0, half);
        v16h bn3 = load_bfrag(B3, k0, half);
        acc[0] = WMMA_F16(af, bf0, acc[0]);
        acc[1] = WMMA_F16(af, bf1, acc[1]);
        acc[2] = WMMA_F16(af, bf2, acc[2]);
        acc[3] = WMMA_F16(af, bf3, acc[3]);
        af = afn; bf0 = bn0; bf1 = bn1; bf2 = bn2; bf3 = bn3;
    }
    acc[0] = WMMA_F16(af, bf0, acc[0]);
    acc[1] = WMMA_F16(af, bf1, acc[1]);
    acc[2] = WMMA_F16(af, bf2, acc[2]);
    acc[3] = WMMA_F16(af, bf3, acc[3]);
}

// ---------------------------------------------------------------------------
// Kernel 1: LayerNorm (f32 in -> f16 out). grid.y==0: query (also emits raw
// f16 copy of x for the gate GEMM); grid.y==1: support.
// ---------------------------------------------------------------------------
__global__ __launch_bounds__(256) void ln_kernel(
    const float* __restrict__ xq, const float* __restrict__ xs,
    const float* __restrict__ wq, const float* __restrict__ bq,
    const float* __restrict__ ws_, const float* __restrict__ bs,
    _Float16* __restrict__ lnq, _Float16* __restrict__ lns,
    _Float16* __restrict__ xq16) {
    const int row = blockIdx.x;
    const int sel = blockIdx.y;
    const float* x = (sel == 0 ? xq : xs) + (size_t)row * CLIP;
    const float* w = sel == 0 ? wq : ws_;
    const float* bb = sel == 0 ? bq : bs;
    _Float16* o = (sel == 0 ? lnq : lns) + (size_t)row * CLIP;

    __shared__ float red[256];
    const int tid = threadIdx.x;
    float4 v4 = ((const float4*)x)[tid];
    float vv[4] = {v4.x, v4.y, v4.z, v4.w};

    float s = vv[0] + vv[1] + vv[2] + vv[3];
    red[tid] = s;
    __syncthreads();
    for (int off = 128; off > 0; off >>= 1) {
        if (tid < off) red[tid] += red[tid + off];
        __syncthreads();
    }
    const float mu = red[0] * (1.0f / CLIP);
    __syncthreads();

    float s2 = 0.f;
    for (int j = 0; j < 4; ++j) { float d = vv[j] - mu; s2 += d * d; }
    red[tid] = s2;
    __syncthreads();
    for (int off = 128; off > 0; off >>= 1) {
        if (tid < off) red[tid] += red[tid + off];
        __syncthreads();
    }
    const float rs = rsqrtf(red[0] * (1.0f / CLIP) + 1e-5f);

    for (int j = 0; j < 4; ++j) {
        int c = tid * 4 + j;
        o[c] = (_Float16)((vv[j] - mu) * rs * w[c] + bb[c]);
        if (sel == 0) xq16[(size_t)row * CLIP + c] = (_Float16)vv[j];
    }
}

// ---------------------------------------------------------------------------
// Kernel 2: weight convert+transpose f32 (KxN) -> f16 (NxK)
// ---------------------------------------------------------------------------
__global__ __launch_bounds__(256) void wt_kernel(const float* __restrict__ src,
                                                 _Float16* __restrict__ dst,
                                                 int Kdim, int Ndim) {
    long idx = (long)blockIdx.x * 256 + threadIdx.x;
    if (idx >= (long)Kdim * Ndim) return;
    int k = (int)(idx / Ndim);
    int n = (int)(idx % Ndim);
    dst[(size_t)n * Kdim + k] = (_Float16)src[idx];
}

// ---------------------------------------------------------------------------
// Kernel 3: QKV projection. X_ln(16384x1024) @ W^T, K=1024. Each wave
// computes a 16x64 strip (4 accumulator tiles, double-buffered fragments).
// sel: 0 = Q (scaled 1/8), 1 = K, 2 = V (head-transposed Vt[b][h][d][n]).
// ---------------------------------------------------------------------------
__global__ __launch_bounds__(256) void proj_kernel(
    const _Float16* __restrict__ lnq, const _Float16* __restrict__ lns,
    const _Float16* __restrict__ WqT, const _Float16* __restrict__ WkT,
    const _Float16* __restrict__ WvT,
    const float* __restrict__ bq, const float* __restrict__ bk,
    const float* __restrict__ bv,
    _Float16* __restrict__ Qf, _Float16* __restrict__ Kf,
    _Float16* __restrict__ Vt) {
    const int tid = threadIdx.x, lane = tid & 31, wv = tid >> 5;
    const int half = lane >> 4, l16 = lane & 15;
    const int task = blockIdx.x * 8 + wv;   // 1024 row-tiles x 4 col-blocks
    const int tm = task >> 2, cb = task & 3;
    const int sel = blockIdx.y;

    const _Float16* A  = (sel == 0) ? lnq : lns;
    const _Float16* BT = (sel == 0) ? WqT : (sel == 1) ? WkT : WvT;
    const float* bias  = (sel == 0) ? bq  : (sel == 1) ? bk  : bv;

    const _Float16* Arow = A + (size_t)(tm * 16 + l16) * CLIP;
    const _Float16* Bbase = BT + (size_t)(cb * 64 + l16) * CLIP;

    v8f acc[4] = {};
    gemm_strip4(Arow, Bbase, Bbase + 16 * CLIP, Bbase + 32 * CLIP, Bbase + 48 * CLIP,
                32, half, acc);

    for (int j = 0; j < 4; ++j) {
        const int colg = (cb * 4 + j) * 16 + l16;   // 0..255
        const float bval = bias[colg];
        if (sel == 2) {
            const int h = colg >> 6, d = colg & 63;
            for (int r = 0; r < 8; ++r) {
                int g = tm * 16 + r + 8 * half;
                int b = g >> 10, n = g & 1023;
                Vt[((size_t)(b * NHEAD + h) * HDIM + d) * NN + n] =
                    (_Float16)(acc[j][r] + bval);
            }
        } else {
            _Float16* O = (sel == 0) ? Qf : Kf;
            const float scale = (sel == 0) ? 0.125f : 1.0f;  // fold 1/sqrt(64) into Q
            for (int r = 0; r < 8; ++r) {
                int g = tm * 16 + r + 8 * half;
                O[(size_t)g * INNER + colg] = (_Float16)((acc[j][r] + bval) * scale);
            }
        }
    }
}

// ---------------------------------------------------------------------------
// Kernel 4: fused attention. One block per (b*h, 16-row tile of queries).
// logits (WMMA, A reused across 8 col tiles) -> +mask bias -> LDS softmax
// -> attn f32 to d_out -> context = attn @ V (WMMA) -> ctx f16 to workspace.
// ---------------------------------------------------------------------------
__global__ __launch_bounds__(256) void attn_kernel(
    const _Float16* __restrict__ Qf, const _Float16* __restrict__ Kf,
    const _Float16* __restrict__ Vt, const float* __restrict__ maskw,
    float* __restrict__ attn_out, _Float16* __restrict__ ctx) {
    __shared__ float    sl[16][NN];     // 64 KB: logits / exp values
    __shared__ _Float16 sa[16][NN];     // 32 KB: attention probs (f16)
    __shared__ float    sred[16][17];

    const int tid = threadIdx.x, lane = tid & 31, wv = tid >> 5;
    const int half = lane >> 4, l16 = lane & 15;
    const int bh = blockIdx.x;          // 0..63
    const int b = bh >> 2, h = bh & 3;
    const int rt = blockIdx.y;          // 0..63 (query row tile)

    // --- Q fragments for d = 0..63 (2 k-steps of 32), per-wave registers
    const _Float16* Qbase = Qf + (size_t)(b * NN + rt * 16 + l16) * INNER + h * HDIM;
    v16h aq0 = load_afrag(Qbase, 0, half);
    v16h aq1 = load_afrag(Qbase, 32, half);

    // --- logits: wave wv owns column tiles wv*8 .. wv*8+7
    for (int i = 0; i < 8; ++i) {
        const int ct = wv * 8 + i;
        const int key = ct * 16 + l16;
        const _Float16* Kp = Kf + (size_t)(b * NN + key) * INNER + h * HDIM;
        v8f acc = {};
        v16h bf0 = load_bfrag(Kp, 0, half);
        v16h bf1 = load_bfrag(Kp, 32, half);
        acc = WMMA_F16(aq0, bf0, acc);
        acc = WMMA_F16(aq1, bf1, acc);
        const float mb = 5.0f * maskw[b * NN + key];
        for (int r = 0; r < 8; ++r)
            sl[r + 8 * half][ct * 16 + l16] = acc[r] + mb;
    }
    __syncthreads();

    // --- softmax: row r handled by 16 threads, 64 columns each
    const int r  = tid >> 4;
    const int g16 = tid & 15;
    const int c0 = g16 << 6;
    float m = -3.4e38f;
    for (int c = 0; c < 64; ++c) m = fmaxf(m, sl[r][c0 + c]);
    sred[r][g16] = m;
    __syncthreads();
    if (g16 == 0) {
        float mm = sred[r][0];
        for (int j = 1; j < 16; ++j) mm = fmaxf(mm, sred[r][j]);
        sred[r][16] = mm;
    }
    __syncthreads();
    m = sred[r][16];
    float ssum = 0.f;
    for (int c = 0; c < 64; ++c) {
        float e = __expf(sl[r][c0 + c] - m);
        sl[r][c0 + c] = e;
        ssum += e;
    }
    sred[r][g16] = ssum;
    __syncthreads();
    if (g16 == 0) {
        float st = 0.f;
        for (int j = 0; j < 16; ++j) st += sred[r][j];
        sred[r][16] = st;
    }
    __syncthreads();
    const float inv = 1.0f / sred[r][16];
    float* arow = attn_out + ((size_t)bh * NN + rt * 16 + r) * NN;
    for (int c = 0; c < 64; ++c) {
        float p = sl[r][c0 + c] * inv;
        arow[c0 + c] = p;                 // attn output (f32)
        sa[r][c0 + c] = (_Float16)p;      // f16 copy for context GEMM
    }
    __syncthreads();

    // --- context: waves 0..3 each compute a 16x16 d-tile, K loop over 1024
    if (wv < 4) {
        const _Float16* Vp = Vt + (size_t)(bh * HDIM + wv * 16 + l16) * NN;
        const _Float16* Ap = &sa[l16][0];
        v8f acc = {};
        v16h af = load_afrag(Ap, 0, half);
        v16h bf = load_bfrag(Vp, 0, half);
        for (int kt = 0; kt < 31; ++kt) {
            v16h afn = load_afrag(Ap, (kt + 1) * 32, half);
            v16h bfn = load_bfrag(Vp, (kt + 1) * 32, half);
            acc = WMMA_F16(af, bf, acc);
            af = afn; bf = bfn;
        }
        acc = WMMA_F16(af, bf, acc);
        for (int rr = 0; rr < 8; ++rr) {
            int row = rt * 16 + rr + 8 * half;
            ctx[(size_t)(b * NN + row) * INNER + h * HDIM + wv * 16 + l16] =
                (_Float16)acc[rr];
        }
    }
}

// ---------------------------------------------------------------------------
// Kernel 5: epilogue. gate = sigmoid(x@Wg+bg) (K=1024), cross = ctx@Wo+bo
// (K=256), enhanced = x + gate*cross. Each wave computes a 16x64 strip
// (4 accumulator tiles, double-buffered fragments) for both GEMMs.
// ---------------------------------------------------------------------------
__global__ __launch_bounds__(256) void out_kernel(
    const float* __restrict__ x, const _Float16* __restrict__ xq16,
    const _Float16* __restrict__ ctx, const _Float16* __restrict__ WgT,
    const _Float16* __restrict__ WoT, const float* __restrict__ bg,
    const float* __restrict__ bo, float* __restrict__ out) {
    const int tid = threadIdx.x, lane = tid & 31, wv = tid >> 5;
    const int half = lane >> 4, l16 = lane & 15;
    const int task = blockIdx.x * 8 + wv;   // 1024 row-tiles x 16 col-blocks
    const int tm = task >> 4, cb = task & 15;

    // gate GEMM: K = 1024
    const _Float16* Arow = xq16 + (size_t)(tm * 16 + l16) * CLIP;
    const _Float16* Bgb = WgT + (size_t)(cb * 64 + l16) * CLIP;
    v8f gacc[4] = {};
    gemm_strip4(Arow, Bgb, Bgb + 16 * CLIP, Bgb + 32 * CLIP, Bgb + 48 * CLIP,
                32, half, gacc);

    // cross GEMM: K = 256
    const _Float16* A2 = ctx + (size_t)(tm * 16 + l16) * INNER;
    const _Float16* Bob = WoT + (size_t)(cb * 64 + l16) * INNER;
    v8f cacc[4] = {};
    gemm_strip4(A2, Bob, Bob + 16 * INNER, Bob + 32 * INNER, Bob + 48 * INNER,
                8, half, cacc);

    for (int j = 0; j < 4; ++j) {
        const int colg = cb * 64 + j * 16 + l16;
        const float bgv = bg[colg], bov = bo[colg];
        for (int rr = 0; rr < 8; ++rr) {
            size_t row = (size_t)(tm * 16 + rr + 8 * half);
            float xv = x[row * CLIP + colg];
            float g = 1.0f / (1.0f + __expf(-(gacc[j][rr] + bgv)));
            out[row * CLIP + colg] = xv + g * (cacc[j][rr] + bov);
        }
    }
}

// ---------------------------------------------------------------------------
extern "C" void kernel_launch(void* const* d_in, const int* in_sizes, int n_in,
                              void* d_out, int out_size, void* d_ws, size_t ws_size,
                              hipStream_t stream) {
    (void)in_sizes; (void)n_in; (void)out_size; (void)ws_size;

    const float* xq    = (const float*)d_in[0];   // query_features
    const float* xs    = (const float*)d_in[1];   // support_features
    const float* maskw = (const float*)d_in[2];   // support_mask_weights
    const float* lnqw  = (const float*)d_in[3];
    const float* lnqb  = (const float*)d_in[4];
    const float* lnsw  = (const float*)d_in[5];
    const float* lnsb  = (const float*)d_in[6];
    const float* Wq = (const float*)d_in[7];  const float* bq = (const float*)d_in[8];
    const float* Wk = (const float*)d_in[9];  const float* bk = (const float*)d_in[10];
    const float* Wv = (const float*)d_in[11]; const float* bv = (const float*)d_in[12];
    const float* Wo = (const float*)d_in[13]; const float* bo = (const float*)d_in[14];
    const float* Wg = (const float*)d_in[15]; const float* bg = (const float*)d_in[16];

    // d_out = [enhanced (B,N,1024) f32][attn (B,H,N,N) f32]
    float* enhanced = (float*)d_out;
    float* attn_out = (float*)d_out + (size_t)NB * NN * CLIP;

    // Workspace layout (bytes)
    char* ws = (char*)d_ws;
    _Float16* lnq16 = (_Float16*)(ws + 0);                       // 32 MB
    _Float16* lns16 = (_Float16*)(ws + 33554432ull);             // 32 MB
    _Float16* xq16  = (_Float16*)(ws + 67108864ull);             // 32 MB
    _Float16* Qf    = (_Float16*)(ws + 100663296ull);            // 8 MB
    _Float16* Kf    = (_Float16*)(ws + 109051904ull);            // 8 MB
    _Float16* Vt    = (_Float16*)(ws + 117440512ull);            // 8 MB
    _Float16* ctx   = (_Float16*)(ws + 125829120ull);            // 8 MB
    _Float16* WqT   = (_Float16*)(ws + 134217728ull);            // 512 KB
    _Float16* WkT   = (_Float16*)(ws + 134742016ull);
    _Float16* WvT   = (_Float16*)(ws + 135266304ull);
    _Float16* WoT   = (_Float16*)(ws + 135790592ull);
    _Float16* WgT   = (_Float16*)(ws + 136314880ull);            // 2 MB

    // 1) LayerNorm (+ raw f16 copy of query features)
    ln_kernel<<<dim3(NROWS, 2), 256, 0, stream>>>(xq, xs, lnqw, lnqb, lnsw, lnsb,
                                                  lnq16, lns16, xq16);
    // 2) weight transpose + f16 convert
    wt_kernel<<<(CLIP * INNER + 255) / 256, 256, 0, stream>>>(Wq, WqT, CLIP, INNER);
    wt_kernel<<<(CLIP * INNER + 255) / 256, 256, 0, stream>>>(Wk, WkT, CLIP, INNER);
    wt_kernel<<<(CLIP * INNER + 255) / 256, 256, 0, stream>>>(Wv, WvT, CLIP, INNER);
    wt_kernel<<<(INNER * CLIP + 255) / 256, 256, 0, stream>>>(Wo, WoT, INNER, CLIP);
    wt_kernel<<<(CLIP * CLIP + 255) / 256, 256, 0, stream>>>(Wg, WgT, CLIP, CLIP);
    // 3) QKV projections (grid.y selects Q/K/V); 4096 wave-tasks per matrix
    proj_kernel<<<dim3(512, 3), 256, 0, stream>>>(lnq16, lns16, WqT, WkT, WvT,
                                                  bq, bk, bv, Qf, Kf, Vt);
    // 4) fused attention
    attn_kernel<<<dim3(NB * NHEAD, NN / 16), 256, 0, stream>>>(Qf, Kf, Vt, maskw,
                                                               attn_out, ctx);
    // 5) gate + cross + residual; 16384 wave-tasks
    out_kernel<<<2048, 256, 0, stream>>>(xq, xq16, ctx, WgT, WoT, bg, bo, enhanced);
}
